// GCNVAE_12799002542722
// MI455X (gfx1250) — matrix operand order repaired and checked by reading, not compiled, not run on hardware
//
#include <hip/hip_runtime.h>
#include <hip/hip_bf16.h>
#include <cstdint>

#define N_   4096
#define D_   256
#define HID_ 256
#define OUT_ 128
#define HEADS_ 4
#define DH_  64
#define E_   131072

typedef __attribute__((ext_vector_type(16))) __bf16 v16bf;
typedef __attribute__((ext_vector_type(8)))  __bf16 v8bf;
typedef __attribute__((ext_vector_type(8)))  float  v8f;

// ---- WMMA helpers -----------------------------------------------------------
// A operand (16x32 bf16, MxK): lane lm = row; elem i -> K = 16*hl + i.
// Two contiguous 16B loads.
__device__ __forceinline__ v16bf load_a16(const __bf16* base, int hl) {
  v8bf a0 = *(const v8bf*)(base + 16 * hl);
  v8bf a1 = *(const v8bf*)(base + 16 * hl + 8);
  v16bf r;
#pragma unroll
  for (int i = 0; i < 8; ++i) { r[i] = a0[i]; r[8 + i] = a1[i]; }
  return r;
}
// B operand (32x16 bf16, KxN): lane lm = col; elem i -> K = i + 16*hl.
__device__ __forceinline__ v8f wmma_bf16(v16bf a, v16bf b, v8f c) {
  return __builtin_amdgcn_wmma_f32_16x16x32_bf16(false, a, false, b, (short)0, c,
                                                 false, false);
}
// C tile (16x16 f32): lane lm = col; vgpr r -> row = r + 8*hl.

// ---- CDNA5 async global->LDS copy (ASYNCcnt-tracked DMA), 16B per lane ------
// Low 32 bits of a generic LDS pointer are the wave-relative LDS byte offset
// (ISA 10.2, LDS aperture).
__device__ __forceinline__ void async_cp16(const void* g, void* lds_generic) {
  uint32_t loff = (uint32_t)(uintptr_t)lds_generic;
  asm volatile("global_load_async_to_lds_b128 %0, %1, off"
               :: "v"(loff), "v"(g) : "memory");
}

// ---- conversion kernels -----------------------------------------------------
__global__ void k_cvt(const float* __restrict__ in, __bf16* __restrict__ out, int n) {
  int i = blockIdx.x * blockDim.x + threadIdx.x;
  if (i < n) out[i] = (__bf16)in[i];
}
// in[R][C] f32 -> out[C][R] bf16
__global__ void k_cvt_t(const float* __restrict__ in, __bf16* __restrict__ out,
                        int R, int C) {
  int i = blockIdx.x * blockDim.x + threadIdx.x;
  if (i < R * C) {
    int r = i / C, c = i - r * C;
    out[c * R + r] = (__bf16)in[i];
  }
}

// ---- QKV projection: qkv = x @ Wqkv^T + bqkv, scatter to per-head q/k/vT ----
__global__ void k_qkv(const __bf16* __restrict__ xb, const __bf16* __restrict__ wq,
                      const float* __restrict__ bqkv, __bf16* __restrict__ qb,
                      __bf16* __restrict__ kb, __bf16* __restrict__ vt) {
  const int lane = threadIdx.x & 31, wv = threadIdx.x >> 5;
  const int lm = lane & 15, hl = lane >> 4;
  const int m0 = blockIdx.x * 128 + wv * 16;
  const int c0 = blockIdx.y * 64;
  v8f acc[4];
#pragma unroll
  for (int jt = 0; jt < 4; ++jt) {
    float bj = bqkv[c0 + jt * 16 + lm];
#pragma unroll
    for (int r = 0; r < 8; ++r) acc[jt][r] = bj;
  }
  const __bf16* arow = xb + (m0 + lm) * D_;
#pragma unroll
  for (int kk = 0; kk < D_; kk += 32) {
    v16bf a = load_a16(arow + kk, hl);
#pragma unroll
    for (int jt = 0; jt < 4; ++jt) {
      v16bf b = *(const v16bf*)(wq + (c0 + jt * 16 + lm) * D_ + kk + 16 * hl);
      acc[jt] = wmma_bf16(a, b, acc[jt]);
    }
  }
#pragma unroll
  for (int jt = 0; jt < 4; ++jt) {
    int j0 = c0 + jt * 16;
    int sect = j0 >> 8;             // 0=q,1=k,2=v (16-col tile never crosses)
    int head = (j0 & 255) >> 6;
    int dc = (j0 + lm) & 63;
#pragma unroll
    for (int r = 0; r < 8; ++r) {
      int row = m0 + r + 8 * hl;
      float v = acc[jt][r];
      if (sect == 0)      qb[((head * N_ + row) << 6) + dc] = (__bf16)(v * 0.125f);
      else if (sect == 1) kb[((head * N_ + row) << 6) + dc] = (__bf16)v;
      else                vt[(head * 64 + dc) * N_ + row] = (__bf16)v;
    }
  }
}

// ---- flash attention: S^T formulation, async double-buffered K/V in LDS -----
__global__ void k_attn(const __bf16* __restrict__ qb, const __bf16* __restrict__ kb,
                       const __bf16* __restrict__ vt, __bf16* __restrict__ attnb) {
  __shared__ __align__(16) __bf16 ksh[2][32 * 64];  // [buf][key][dh]   8 KB
  __shared__ __align__(16) __bf16 vsh[2][64 * 32];  // [buf][dh][key]   8 KB
  const int t = threadIdx.x;
  const int lane = t & 31, wv = t >> 5;
  const int lm = lane & 15, hl = lane >> 4;
  const int h = blockIdx.y;
  const int q0 = blockIdx.x * 128 + wv * 16;
  const char* kg = (const char*)(kb + (size_t)h * N_ * 64);

  v16bf qB[2];   // B operand of S^T = K * Q^T : col = query, K = dh
#pragma unroll
  for (int b = 0; b < 2; ++b)
    qB[b] = *(const v16bf*)(qb + ((h * N_ + q0 + lm) << 6) + b * 32 + 16 * hl);

  v8f acc[4] = {v8f{}, v8f{}, v8f{}, v8f{}};   // out^T accum: 64 dh rows x 16 q
  float m_run = -1.0e30f, l_run = 0.f;

  // issue one chunk's DMA: 2 async b128 per thread (K 4KB contig + V^T rows)
  const int vm = t >> 2, vo = (t & 3) * 16;          // V^T: 4 threads per row
  const char* vg = (const char*)(vt + (size_t)(h * 64 + vm) * N_) + vo;
#define ISSUE(c)                                                          \
  do {                                                                    \
    int buf_ = (c) & 1;                                                   \
    int key0_ = (c) * 32;                                                 \
    async_cp16(kg + (size_t)key0_ * 128 + t * 16, &ksh[buf_][t * 8]);     \
    async_cp16(vg + (size_t)key0_ * 2, &vsh[buf_][vm * 32 + (t & 3) * 8]);\
  } while (0)

  ISSUE(0);
  const int NC = N_ / 32;
  for (int c = 0; c < NC; ++c) {
    if (c + 1 < NC) {
      ISSUE(c + 1);
      asm volatile("s_wait_asynccnt 0x2" ::: "memory");  // chunk c done (in-order)
    } else {
      asm volatile("s_wait_asynccnt 0x0" ::: "memory");
    }
    __syncthreads();
    const int buf = c & 1;

    const __bf16* kp0 = &ksh[buf][lm * 64];
    const __bf16* kp1 = &ksh[buf][(16 + lm) * 64];
    v8f st0 = {}, st1 = {};
    st0 = wmma_bf16(load_a16(kp0, hl), qB[0], st0);
    st0 = wmma_bf16(load_a16(kp0 + 32, hl), qB[1], st0);
    st1 = wmma_bf16(load_a16(kp1, hl), qB[0], st1);
    st1 = wmma_bf16(load_a16(kp1 + 32, hl), qB[1], st1);

    // online softmax over keys (register dim + half-wave partner)
    float mc = -1.0e30f;
#pragma unroll
    for (int r = 0; r < 8; ++r) { mc = fmaxf(mc, st0[r]); mc = fmaxf(mc, st1[r]); }
    mc = fmaxf(mc, __shfl_xor(mc, 16, 32));
    float m_new = fmaxf(m_run, mc);
    float scale = __expf(m_run - m_new);
    float p0[8], p1[8], lc = 0.f;
#pragma unroll
    for (int r = 0; r < 8; ++r) {
      p0[r] = __expf(st0[r] - m_new);
      p1[r] = __expf(st1[r] - m_new);
      lc += p0[r] + p1[r];
    }
    lc += __shfl_xor(lc, 16, 32);
    l_run = l_run * scale + lc;
    m_run = m_new;
#pragma unroll
    for (int mt = 0; mt < 4; ++mt)
#pragma unroll
      for (int r = 0; r < 8; ++r) acc[mt][r] *= scale;

    // C tile (key=r+8*hl) -> B operand (key=i+16*hl): swap halves via shfl
    v16bf pB;
#pragma unroll
    for (int r = 0; r < 8; ++r) {
      float t0o = __shfl_xor(p0[r], 16, 32);
      float t1o = __shfl_xor(p1[r], 16, 32);
      pB[r]     = (__bf16)(hl ? t1o : p0[r]);
      pB[8 + r] = (__bf16)(hl ? p1[r] : t0o);
    }
    // out^T += V^T(64 x 32keys) * P^T(32keys x 16q); V staged transposed
#pragma unroll
    for (int mt = 0; mt < 4; ++mt) {
      const __bf16* vp = &vsh[buf][(mt * 16 + lm) * 32];
      acc[mt] = wmma_bf16(load_a16(vp, hl), pB, acc[mt]);
    }
    __syncthreads();   // all waves done reading buf before it is re-filled
  }
#undef ISSUE

  float inv = 1.f / l_run;
  int q = q0 + lm;
#pragma unroll
  for (int mt = 0; mt < 4; ++mt)
#pragma unroll
    for (int r = 0; r < 8; ++r) {
      int dhr = mt * 16 + r + 8 * hl;
      attnb[q * D_ + h * 64 + dhr] = (__bf16)(acc[mt][r] * inv);
    }
}

// ---- generic bf16 WMMA GEMM: C = A[MxK] * B^T-layout[N][K] (+bias) ----------
template <bool HASBIAS, typename OutT>
__global__ void k_gemm(const __bf16* __restrict__ A, const __bf16* __restrict__ B,
                       const float* __restrict__ bias, OutT* __restrict__ Cout,
                       int K, int Ncols) {
  const int lane = threadIdx.x & 31, wv = threadIdx.x >> 5;
  const int lm = lane & 15, hl = lane >> 4;
  const int m0 = blockIdx.x * 128 + wv * 16;
  const int c0 = blockIdx.y * 64;
  v8f acc[4];
#pragma unroll
  for (int jt = 0; jt < 4; ++jt) {
    float bj = HASBIAS ? bias[c0 + jt * 16 + lm] : 0.f;
#pragma unroll
    for (int r = 0; r < 8; ++r) acc[jt][r] = bj;
  }
  const __bf16* arow = A + (m0 + lm) * K;
  for (int kk = 0; kk < K; kk += 32) {
    v16bf a = load_a16(arow + kk, hl);
#pragma unroll
    for (int jt = 0; jt < 4; ++jt) {
      v16bf b = *(const v16bf*)(B + (c0 + jt * 16 + lm) * K + kk + 16 * hl);
      acc[jt] = wmma_bf16(a, b, acc[jt]);
    }
  }
#pragma unroll
  for (int jt = 0; jt < 4; ++jt) {
    int j = c0 + jt * 16 + lm;
#pragma unroll
    for (int r = 0; r < 8; ++r) {
      int row = m0 + r + 8 * hl;
      Cout[row * Ncols + j] = (OutT)acc[jt][r];
    }
  }
}

// ---- GCN aggregation helpers ------------------------------------------------
__global__ void k_deg_init(float* __restrict__ deg) {
  int i = blockIdx.x * blockDim.x + threadIdx.x;
  if (i < N_) deg[i] = 1.0f;   // self-loop
}
__global__ void k_deg_cnt(const int* __restrict__ ei, float* __restrict__ deg) {
  int e = blockIdx.x * blockDim.x + threadIdx.x;
  if (e < E_) atomicAdd(&deg[ei[E_ + e]], 1.0f);
}
// agg = xw * (1/deg) + bias   (self-loop message + bias, init for edge adds)
__global__ void k_self(const float* __restrict__ xw, const float* __restrict__ deg,
                       const float* __restrict__ b, float* __restrict__ agg) {
  int i = blockIdx.x * blockDim.x + threadIdx.x;
  if (i < N_ * HID_) {
    int row = i >> 8;
    agg[i] = xw[i] * (1.0f / deg[row]) + b[i & (HID_ - 1)];
  }
}
// one block per edge, one thread per feature
__global__ void k_edge_agg(const int* __restrict__ ei, const float* __restrict__ xw,
                           const float* __restrict__ deg, float* __restrict__ agg) {
  int e = blockIdx.x;
  int s = ei[e], d = ei[E_ + e];
  float norm = rsqrtf(deg[s]) * rsqrtf(deg[d]);
  int f = threadIdx.x;
  atomicAdd(&agg[d * HID_ + f], xw[s * HID_ + f] * norm);
}
__global__ void k_act(const float* __restrict__ in, __bf16* __restrict__ out,
                      int n, int relu) {
  int i = blockIdx.x * blockDim.x + threadIdx.x;
  if (i < n) {
    float v = in[i];
    out[i] = (__bf16)(relu ? fmaxf(v, 0.f) : v);
  }
}

// ---- fused mu / logvar GEMMs + reparameterization; writes (z, mu, z) --------
__global__ void k_muz(const __bf16* __restrict__ A, const __bf16* __restrict__ Bmu,
                      const __bf16* __restrict__ Blv, const float* __restrict__ bmu,
                      const float* __restrict__ blv, const float* __restrict__ eps,
                      float* __restrict__ out) {
  const int lane = threadIdx.x & 31, wv = threadIdx.x >> 5;
  const int lm = lane & 15, hl = lane >> 4;
  const int m0 = blockIdx.x * 16;
  const int j0 = wv * 16;                      // 8 waves cover OUT_=128 cols
  v8f amu, alv;
  float bm = bmu[j0 + lm], bl = blv[j0 + lm];
#pragma unroll
  for (int r = 0; r < 8; ++r) { amu[r] = bm; alv[r] = bl; }
  const __bf16* arow = A + (m0 + lm) * HID_;
#pragma unroll
  for (int kk = 0; kk < HID_; kk += 32) {
    v16bf a = load_a16(arow + kk, hl);
    v16bf bm16 = *(const v16bf*)(Bmu + (j0 + lm) * HID_ + kk + 16 * hl);
    v16bf bl16 = *(const v16bf*)(Blv + (j0 + lm) * HID_ + kk + 16 * hl);
    amu = wmma_bf16(a, bm16, amu);
    alv = wmma_bf16(a, bl16, alv);
  }
  int j = j0 + lm;
#pragma unroll
  for (int r = 0; r < 8; ++r) {
    int row = m0 + r + 8 * hl;
    float mu = amu[r], lv = alv[r];
    float e = eps[row * OUT_ + j];
    float z = mu + e * __expf(0.5f * lv);
    out[row * OUT_ + j] = z;
    out[N_ * OUT_ + row * OUT_ + j] = mu;
    out[2 * N_ * OUT_ + row * OUT_ + j] = z;
  }
}

// ---- host-side orchestration ------------------------------------------------
extern "C" void kernel_launch(void* const* d_in, const int* in_sizes, int n_in,
                              void* d_out, int out_size, void* d_ws, size_t ws_size,
                              hipStream_t stream) {
  (void)in_sizes; (void)n_in; (void)out_size; (void)ws_size;
  const float* x    = (const float*)d_in[0];
  const int*   ei   = (const int*)d_in[1];
  const float* eps  = (const float*)d_in[2];
  const float* Wqkv = (const float*)d_in[3];
  const float* bqkv = (const float*)d_in[4];
  const float* Wo   = (const float*)d_in[5];
  const float* bo   = (const float*)d_in[6];
  const float* W1   = (const float*)d_in[7];
  const float* b1   = (const float*)d_in[8];
  const float* W2   = (const float*)d_in[9];
  const float* b2   = (const float*)d_in[10];
  const float* Wmu  = (const float*)d_in[11];
  const float* bmu  = (const float*)d_in[12];
  const float* Wlv  = (const float*)d_in[13];
  const float* blv  = (const float*)d_in[14];
  float* out = (float*)d_out;

  char* ws = (char*)d_ws;
  size_t off = 0;
  auto alloc = [&](size_t bytes) -> char* {
    char* p = ws + off;
    off += (bytes + 255) & ~size_t(255);
    return p;
  };
  __bf16* xb    = (__bf16*)alloc((size_t)N_ * D_ * 2);
  __bf16* wqkvb = (__bf16*)alloc((size_t)3 * D_ * D_ * 2);
  __bf16* wob   = (__bf16*)alloc((size_t)D_ * D_ * 2);
  __bf16* w1t   = (__bf16*)alloc((size_t)D_ * HID_ * 2);
  __bf16* w2t   = (__bf16*)alloc((size_t)HID_ * HID_ * 2);
  __bf16* wmut  = (__bf16*)alloc((size_t)HID_ * OUT_ * 2);
  __bf16* wlvt  = (__bf16*)alloc((size_t)HID_ * OUT_ * 2);
  __bf16* qbuf  = (__bf16*)alloc((size_t)HEADS_ * N_ * DH_ * 2);
  __bf16* kbuf  = (__bf16*)alloc((size_t)HEADS_ * N_ * DH_ * 2);
  __bf16* vt    = (__bf16*)alloc((size_t)HEADS_ * N_ * DH_ * 2);
  __bf16* attnb = (__bf16*)alloc((size_t)N_ * D_ * 2);
  __bf16* hb    = (__bf16*)alloc((size_t)N_ * D_ * 2);
  __bf16* h1b   = (__bf16*)alloc((size_t)N_ * HID_ * 2);
  __bf16* h2b   = (__bf16*)alloc((size_t)N_ * HID_ * 2);
  float*  xw    = (float*)alloc((size_t)N_ * HID_ * 4);
  float*  agg   = (float*)alloc((size_t)N_ * HID_ * 4);
  float*  deg   = (float*)alloc((size_t)N_ * 4);

  dim3 B(256);
  // precision conversion / weight re-layout
  k_cvt<<<(N_ * D_) / 256, B, 0, stream>>>(x, xb, N_ * D_);
  k_cvt<<<(3 * D_ * D_) / 256, B, 0, stream>>>(Wqkv, wqkvb, 3 * D_ * D_);
  k_cvt<<<(D_ * D_) / 256, B, 0, stream>>>(Wo, wob, D_ * D_);
  k_cvt_t<<<(D_ * HID_) / 256, B, 0, stream>>>(W1, w1t, D_, HID_);
  k_cvt_t<<<(HID_ * HID_) / 256, B, 0, stream>>>(W2, w2t, HID_, HID_);
  k_cvt_t<<<(HID_ * OUT_) / 256, B, 0, stream>>>(Wmu, wmut, HID_, OUT_);
  k_cvt_t<<<(HID_ * OUT_) / 256, B, 0, stream>>>(Wlv, wlvt, HID_, OUT_);
  // MHA
  k_qkv<<<dim3(N_ / 128, (3 * D_) / 64), B, 0, stream>>>(xb, wqkvb, bqkv, qbuf,
                                                          kbuf, vt);
  k_attn<<<dim3(N_ / 128, HEADS_), B, 0, stream>>>(qbuf, kbuf, vt, attnb);
  k_gemm<true, __bf16><<<dim3(N_ / 128, D_ / 64), B, 0, stream>>>(attnb, wob, bo,
                                                                  hb, D_, D_);
  // GCN layer 1
  k_gemm<false, float><<<dim3(N_ / 128, HID_ / 64), B, 0, stream>>>(hb, w1t,
                                                                    nullptr, xw,
                                                                    D_, HID_);
  k_deg_init<<<N_ / 256, B, 0, stream>>>(deg);
  k_deg_cnt<<<E_ / 256, B, 0, stream>>>(ei, deg);
  k_self<<<(N_ * HID_) / 256, B, 0, stream>>>(xw, deg, b1, agg);
  k_edge_agg<<<E_, B, 0, stream>>>(ei, xw, deg, agg);
  k_act<<<(N_ * HID_) / 256, B, 0, stream>>>(agg, h1b, N_ * HID_, 1);
  // GCN layer 2
  k_gemm<false, float><<<dim3(N_ / 128, HID_ / 64), B, 0, stream>>>(h1b, w2t,
                                                                    nullptr, xw,
                                                                    HID_, HID_);
  k_self<<<(N_ * HID_) / 256, B, 0, stream>>>(xw, deg, b2, agg);
  k_edge_agg<<<E_, B, 0, stream>>>(ei, xw, deg, agg);
  k_act<<<(N_ * HID_) / 256, B, 0, stream>>>(agg, h2b, N_ * HID_, 0);
  // heads + reparameterization -> (z, mu, z)
  k_muz<<<N_ / 16, B, 0, stream>>>(h2b, wmut, wlvt, bmu, blv, eps, out);
}
